// PinSageConv_3977139716599
// MI455X (gfx1250) — matrix operand order repaired and checked by reading, not compiled, not run on hardware
//
#include <hip/hip_runtime.h>
#include <hip/hip_bf16.h>

typedef __attribute__((ext_vector_type(16))) __bf16 v16bf;
typedef __attribute__((ext_vector_type(8)))  float  v8f;

// round-to-nearest-even f32 -> bf16 (used only in one-pass prep kernels)
__device__ __forceinline__ __bf16 to_bf16(float f) {
    unsigned u = __builtin_bit_cast(unsigned, f);
    u += 0x7FFFu + ((u >> 16) & 1u);
    unsigned short hbits = (unsigned short)(u >> 16);
    return __builtin_bit_cast(__bf16, hbits);
}

__global__ void zero_kernel(float* __restrict__ p, long n) {
    long i = (long)blockIdx.x * blockDim.x + threadIdx.x;
    long stride = (long)gridDim.x * blockDim.x;
    for (; i < n; i += stride) p[i] = 0.0f;
}

// plain elementwise f32 -> bf16 (for weight matrices, tiny)
__global__ void cvt_rowmajor_kernel(const float* __restrict__ src,
                                    __bf16* __restrict__ dst, int n) {
    int i = blockIdx.x * blockDim.x + threadIdx.x;
    if (i < n) dst[i] = to_bf16(src[i]);
}

// f32 [N][128] row-major  ->  bf16 in WMMA-A swizzled layout:
// chunk(m, kk, hi) = 16 bf16 at dst + ((m*4+kk)*2+hi)*16, holding
// src[m][kk*32 + hi*16 + 0..15].
// If wsum != nullptr, values are scaled by safediv reciprocal of wsum[m].
__global__ void cvtA_kernel(const float* __restrict__ src,
                            const float* __restrict__ wsum,
                            __bf16* __restrict__ dst, int nNodes) {
    const int idx = blockIdx.x * blockDim.x + threadIdx.x;   // one 32B chunk each
    if (idx >= nNodes * 8) return;
    const int m  = idx >> 3;
    const int kk = (idx >> 1) & 3;
    const int hi = idx & 1;
    float s = 1.0f;
    if (wsum) {
        const float wm = wsum[m];
        s = (wm == 0.0f) ? 1.0f : (1.0f / wm);               // safediv
    }
    const float* base = src + (size_t)m * 128 + kk * 32 + hi * 16;
    v16bf o;
#pragma unroll
    for (int j = 0; j < 8; ++j) {
        o[j]     = to_bf16(base[j] * s);
        o[8 + j] = to_bf16(base[8 + j] * s);
    }
    *(v16bf*)(dst + (size_t)idx * 16) = o;
}

// hq = h @ Qw^T + Qb   (M=nNodes, K=128, N=128); bf16 WMMA, f32 accum.
// hA: swizzled bf16 A operand; QwB: row-major bf16 [128][128].
__global__ void __launch_bounds__(256, 1) gemm1_kernel(const __bf16* __restrict__ hA,
                                                       const __bf16* __restrict__ QwB,
                                                       const float* __restrict__ Qb,
                                                       float* __restrict__ hq,
                                                       int nNodes) {
    const int lane = threadIdx.x & 31;
    const int wave = threadIdx.x >> 5;
    const int tile = blockIdx.x * 8 + wave;     // wave-uniform
    const int m0 = tile * 16;
    if (m0 + 16 > nNodes) return;               // wave-uniform exit, EXEC stays full

    const int lm  = lane & 15;
    const int hi  = (lane >> 4) & 1;
    const int m   = m0 + lm;
    const int klo = hi * 16;

    auto loadA = [&](int kk) -> v16bf {
        return *(const v16bf*)(hA + ((((size_t)m * 4 + kk) * 2 + hi) * 16));
    };
    auto loadB = [&](int kk, int t) -> v16bf {
        return *(const v16bf*)(QwB + (size_t)(t * 16 + lm) * 128 + kk * 32 + klo);
    };

    v8f acc[8];
#pragma unroll
    for (int t = 0; t < 8; ++t) acc[t] = (v8f){0,0,0,0,0,0,0,0};

    // depth-1 software pipeline: B(t+1)/A(kk+1) in flight while WMMA(t) issues
    v16bf a = loadA(0);
#pragma unroll
    for (int kk = 0; kk < 4; ++kk) {            // K = 128 in steps of 32
        v16bf anext;
        if (kk < 3) anext = loadA(kk + 1);
        v16bf bcur = loadB(kk, 0);
#pragma unroll
        for (int t = 0; t < 8; ++t) {
            v16bf bnext;
            if (t < 7) bnext = loadB(kk, t + 1);
            acc[t] = __builtin_amdgcn_wmma_f32_16x16x32_bf16(
                false, a, false, bcur, (short)0, acc[t], false, false);
            if (t < 7) bcur = bnext;
        }
        if (kk < 3) a = anext;
    }

    const int ho = hi * 8;
#pragma unroll
    for (int t = 0; t < 8; ++t) {
        const int n = t * 16 + lm;
        const float qb = Qb[n];
#pragma unroll
        for (int v = 0; v < 8; ++v) {
            hq[(size_t)(m0 + v + ho) * 128 + n] = acc[t][v] + qb;
        }
    }
}

// one wave per edge: gather hq[src] row (float4/lane), weighted atomic scatter
__global__ void __launch_bounds__(256) edge_kernel(const float* __restrict__ hq,
                                                   const float* __restrict__ pw,
                                                   const int* __restrict__ src,
                                                   const int* __restrict__ dst,
                                                   float* __restrict__ hagg,
                                                   float* __restrict__ wsum,
                                                   int nEdges) {
    const int lane = threadIdx.x & 31;
    const int e = blockIdx.x * 8 + (threadIdx.x >> 5);   // wave-uniform
    if (e >= nEdges) return;
    const int   s = src[e];
    const int   d = dst[e];
    const float w = pw[e];
    const float4 v = *(const float4*)(hq + (size_t)s * 128 + lane * 4);
    float* outp = hagg + (size_t)d * 128 + lane * 4;
    atomicAdd(outp + 0, v.x * w);
    atomicAdd(outp + 1, v.y * w);
    atomicAdd(outp + 2, v.z * w);
    atomicAdd(outp + 3, v.w * w);
    if (lane == 0) atomicAdd(wsum + d, w);
}

// out = l2norm(leaky_relu([h, hagg/w] @ Ww^T + Wb))  (M=nNodes, K=256, N=128)
// hA / gA: swizzled bf16 A operands (gA already scaled by 1/w); WwB: bf16 [128][256].
__global__ void __launch_bounds__(256, 1) gemm2_kernel(const __bf16* __restrict__ hA,
                                                       const __bf16* __restrict__ gA,
                                                       const __bf16* __restrict__ WwB,
                                                       const float* __restrict__ Wb,
                                                       float* __restrict__ out,
                                                       int nNodes) {
    const int lane = threadIdx.x & 31;
    const int wave = threadIdx.x >> 5;
    const int tile = blockIdx.x * 8 + wave;
    const int m0 = tile * 16;
    if (m0 + 16 > nNodes) return;               // wave-uniform

    const int lm  = lane & 15;
    const int hi  = (lane >> 4) & 1;
    const int m   = m0 + lm;
    const int klo = hi * 16;

    auto loadA = [&](int kk) -> v16bf {
        const __bf16* s = (kk < 4) ? hA : gA;
        return *(const v16bf*)(s + ((((size_t)m * 4 + (kk & 3)) * 2 + hi) * 16));
    };
    auto loadB = [&](int kk, int t) -> v16bf {
        return *(const v16bf*)(WwB + (size_t)(t * 16 + lm) * 256 + kk * 32 + klo);
    };

    v8f acc[8];
#pragma unroll
    for (int t = 0; t < 8; ++t) acc[t] = (v8f){0,0,0,0,0,0,0,0};

    v16bf a = loadA(0);
#pragma unroll
    for (int kk = 0; kk < 8; ++kk) {            // K = 256 in steps of 32
        v16bf anext;
        if (kk < 7) anext = loadA(kk + 1);
        v16bf bcur = loadB(kk, 0);
#pragma unroll
        for (int t = 0; t < 8; ++t) {
            v16bf bnext;
            if (t < 7) bnext = loadB(kk, t + 1);
            acc[t] = __builtin_amdgcn_wmma_f32_16x16x32_bf16(
                false, a, false, bcur, (short)0, acc[t], false, false);
            if (t < 7) bcur = bnext;
        }
        if (kk < 7) a = anext;
    }

    // bias + leaky_relu, per-row sum of squares
    float ss[8];
#pragma unroll
    for (int v = 0; v < 8; ++v) ss[v] = 0.0f;
#pragma unroll
    for (int t = 0; t < 8; ++t) {
        const float wb = Wb[t * 16 + lm];
#pragma unroll
        for (int v = 0; v < 8; ++v) {
            float x = acc[t][v] + wb;
            x = (x > 0.0f) ? x : 0.01f * x;
            acc[t][v] = x;
            ss[v] += x * x;
        }
    }
    // reduce across the 16 lanes of each half-wave (a row lives in one half)
#pragma unroll
    for (int v = 0; v < 8; ++v) {
        float s = ss[v];
        s += __shfl_xor(s, 1, 32);
        s += __shfl_xor(s, 2, 32);
        s += __shfl_xor(s, 4, 32);
        s += __shfl_xor(s, 8, 32);
        const float nrm = sqrtf(s);
        ss[v] = (nrm == 0.0f) ? 1.0f : (1.0f / nrm);   // safediv
    }
    const int ho = hi * 8;
#pragma unroll
    for (int t = 0; t < 8; ++t) {
        const int n = t * 16 + lm;
#pragma unroll
        for (int v = 0; v < 8; ++v) {
            out[(size_t)(m0 + v + ho) * 128 + n] = acc[t][v] * ss[v];
        }
    }
}

extern "C" void kernel_launch(void* const* d_in, const int* in_sizes, int n_in,
                              void* d_out, int out_size, void* d_ws, size_t ws_size,
                              hipStream_t stream) {
    const float* h   = (const float*)d_in[0];
    const float* pw  = (const float*)d_in[1];
    const float* Qw  = (const float*)d_in[2];
    const float* Qb  = (const float*)d_in[3];
    const float* Ww  = (const float*)d_in[4];
    const float* Wb  = (const float*)d_in[5];
    const int*   src = (const int*)d_in[6];
    const int*   dst = (const int*)d_in[7];
    float* out = (float*)d_out;

    const int N = in_sizes[0] / 128;   // nodes
    const int E = in_sizes[1];         // edges

    // workspace layout (all offsets 32B-aligned):
    // hq f32[N*128] | hagg f32[N*128] | wsum f32[N] |
    // hA bf16[N*128] | gA bf16[N*128] | QwB bf16[128*128] | WwB bf16[128*256]
    float*  hq   = (float*)d_ws;
    float*  hagg = hq + (size_t)N * 128;
    float*  wsum = hagg + (size_t)N * 128;
    __bf16* hA   = (__bf16*)(wsum + N);
    __bf16* gA   = hA + (size_t)N * 128;
    __bf16* QwB  = gA + (size_t)N * 128;
    __bf16* WwB  = QwB + 128 * 128;

    // zero scatter accumulators every call (ws is not re-poisoned between replays)
    const long nz = (long)N * 128 + N;          // hagg + wsum are contiguous
    zero_kernel<<<2048, 256, 0, stream>>>(hagg, nz);

    // one-pass bf16 prep (weights + swizzled A operand for h)
    cvt_rowmajor_kernel<<<(128 * 128 + 255) / 256, 256, 0, stream>>>(Qw, QwB, 128 * 128);
    cvt_rowmajor_kernel<<<(128 * 256 + 255) / 256, 256, 0, stream>>>(Ww, WwB, 128 * 256);
    cvtA_kernel<<<(N * 8 + 255) / 256, 256, 0, stream>>>(h, nullptr, hA, N);

    const int tiles = (N + 15) / 16;
    const int gemmBlocks = (tiles + 7) / 8;     // 8 wave-tiles per 256-thread block
    gemm1_kernel<<<gemmBlocks, 256, 0, stream>>>(hA, QwB, Qb, hq, N);

    edge_kernel<<<(E + 7) / 8, 256, 0, stream>>>(hq, pw, src, dst, hagg, wsum, E);

    // fold safediv(., wsum) into the bf16 A-operand conversion of hagg
    cvtA_kernel<<<(N * 8 + 255) / 256, 256, 0, stream>>>(hagg, wsum, gA, N);

    gemm2_kernel<<<gemmBlocks, 256, 0, stream>>>(hA, gA, WwB, Wb, out, N);
}
